// ClassedProjectedAdaptiveLogSoftmax_74852690034770
// MI455X (gfx1250) — matrix operand order repaired and checked by reading, not compiled, not run on hardware
//
#include <hip/hip_runtime.h>
#include <hip/hip_bf16.h>
#include <stdint.h>

typedef __bf16 v16bf __attribute__((ext_vector_type(16)));
typedef float  v8f   __attribute__((ext_vector_type(8)));
typedef unsigned int u32;

#define DD 512          // hidden dim
#define BM 128          // rows per workgroup
#define BN 128          // cols per workgroup
#define KC 128          // K chunk staged in LDS
#define LDT 68          // padded LDS row stride in dwords (KC/2 + 4)
#define NEG_INF (-__builtin_inff())

union FragU { uint4 u[2]; v16bf v; };

// pack two fp32 -> two bf16 (round-to-nearest-even) in one dword
__device__ __forceinline__ u32 pk2bf(float a, float b) {
  u32 ua = __float_as_uint(a), ub = __float_as_uint(b);
  ua += 0x7fffu + ((ua >> 16) & 1u);
  ub += 0x7fffu + ((ub >> 16) & 1u);
  return (ua >> 16) | (ub & 0xffff0000u);
}

// ---------------------------------------------------------------------------
// Fused GEMM + per-64-col-slab online logsumexp partials, one cluster per call
// partial[t * nrows + row] = (rowMax, rowSumExp) over the 64 columns of slab t
// ---------------------------------------------------------------------------
__global__ __launch_bounds__(256) void gemm_lse_kernel(
    const float* __restrict__ H,   // [nrows, 512]
    const float* __restrict__ W,   // weight + clusterStart*512
    const float* __restrict__ B,   // bias + clusterStart
    int ncols, int nrows,
    int r0, int r1, int r2,        // masked (root) columns, relative; -1 if unused
    int tbase,                     // global 64-col slab index base for this cluster
    float2* __restrict__ partial)
{
  __shared__ u32 lA[BM * LDT];
  __shared__ u32 lB[BN * LDT];

  const int tid    = threadIdx.x;
  const int lane   = tid & 31;
  const int waveId = tid >> 5;
  const int half   = (lane >> 4) & 1;
  const int mr     = lane & 15;
  const int wr     = waveId & 3;       // 4 row-waves
  const int wc     = waveId >> 2;      // 2 col-waves
  const int rowBase = wr * 32;
  const int colBase = wc * 64;
  const int row0 = blockIdx.x * BM;
  const int col0 = blockIdx.y * BN;

  v8f acc[2][4] = {};

  for (int kb = 0; kb < DD; kb += KC) {
    // stage A (hidden) and B (weight rows) chunks as bf16 into LDS
    for (int idx = tid; idx < BM * (KC / 4); idx += 256) {
      int r = idx >> 5;                // KC/4 == 32 float4 per row
      int q = idx & 31;
      int grow = row0 + r;
      float4 f = make_float4(0.f, 0.f, 0.f, 0.f);
      if (grow < nrows) f = *(const float4*)(H + (size_t)grow * DD + kb + q * 4);
      lA[r * LDT + q * 2]     = pk2bf(f.x, f.y);
      lA[r * LDT + q * 2 + 1] = pk2bf(f.z, f.w);
      int gcol = col0 + r;
      float4 g = make_float4(0.f, 0.f, 0.f, 0.f);
      if (gcol < ncols) g = *(const float4*)(W + (size_t)gcol * DD + kb + q * 4);
      lB[r * LDT + q * 2]     = pk2bf(g.x, g.y);
      lB[r * LDT + q * 2 + 1] = pk2bf(g.z, g.w);
    }
    __syncthreads();

    #pragma unroll
    for (int ks = 0; ks < KC / 32; ks++) {
      const int kc2 = ks * 16;  // dword offset of this K=32 step
      FragU fa[2];
      // 16-bit A 16x32 layout: lanes<16 K0..7 / K16..23, lanes>=16 K8..15 / K24..31
      #pragma unroll
      for (int rt = 0; rt < 2; rt++) {
        const u32* pa = &lA[(rowBase + rt * 16 + mr) * LDT + kc2 + 4 * half];
        fa[rt].u[0] = *(const uint4*)pa;
        fa[rt].u[1] = *(const uint4*)(pa + 8);
      }
      #pragma unroll
      for (int ct = 0; ct < 4; ct++) {
        // B 32x16 layout: lanes<16 K0..7 / K16..23, lanes>=16 K8..15 / K24..31 (column = lane&15)
        const u32* pb = &lB[(colBase + ct * 16 + mr) * LDT + kc2 + 4 * half];
        FragU fb;
        fb.u[0] = *(const uint4*)pb;
        fb.u[1] = *(const uint4*)(pb + 8);
        #pragma unroll
        for (int rt = 0; rt < 2; rt++) {
          acc[rt][ct] = __builtin_amdgcn_wmma_f32_16x16x32_bf16(
              false, fa[rt].v, false, fb.v, (short)0, acc[rt][ct], false, false);
        }
      }
    }
    __syncthreads();
  }

  // epilogue: bias + root/OOB masking + per-row online (max, sumexp) over 64 cols
  const int tIdx = tbase + blockIdx.y * 2 + wc;
  #pragma unroll
  for (int rt = 0; rt < 2; rt++) {
    float m[8], s[8];
    #pragma unroll
    for (int r = 0; r < 8; r++) { m[r] = NEG_INF; s[r] = 0.f; }
    #pragma unroll
    for (int ct = 0; ct < 4; ct++) {
      int cl = col0 + colBase + ct * 16 + mr;
      bool ok  = (cl < ncols) && (cl != r0) && (cl != r1) && (cl != r2);
      float bv = (cl < ncols) ? B[cl] : 0.f;
      #pragma unroll
      for (int r = 0; r < 8; r++) {
        // C/D layout: VGPR r -> row (r | r+8), column = lane&15
        float v = ok ? (acc[rt][ct][r] + bv) : NEG_INF;
        float mx = v;
        mx = fmaxf(mx, __shfl_xor(mx, 1, 16));
        mx = fmaxf(mx, __shfl_xor(mx, 2, 16));
        mx = fmaxf(mx, __shfl_xor(mx, 4, 16));
        mx = fmaxf(mx, __shfl_xor(mx, 8, 16));
        float e = (v > NEG_INF) ? __expf(v - mx) : 0.f;
        e += __shfl_xor(e, 1, 16);
        e += __shfl_xor(e, 2, 16);
        e += __shfl_xor(e, 4, 16);
        e += __shfl_xor(e, 8, 16);
        float mn = fmaxf(m[r], mx);
        float sn = 0.f;
        if (s[r] > 0.f) sn += s[r] * __expf(m[r] - mn);
        if (e    > 0.f) sn += e    * __expf(mx   - mn);
        m[r] = mn; s[r] = sn;
      }
    }
    #pragma unroll
    for (int r = 0; r < 8; r++) {
      if (mr == r) {
        int grow = row0 + rowBase + rt * 16 + r + 8 * half;
        if (grow < nrows)
          partial[(size_t)tIdx * nrows + grow] = make_float2(m[r], s[r]);
      }
    }
  }
}

// ---------------------------------------------------------------------------
// fp32 wave-per-row dot products: target logit + 3 head cluster logits
// ---------------------------------------------------------------------------
__global__ __launch_bounds__(256) void dots_kernel(
    const float* __restrict__ H, const float* __restrict__ W,
    const float* __restrict__ B, const float* __restrict__ CW,
    const float* __restrict__ CB, const int* __restrict__ target,
    float* __restrict__ targetLogit, float* __restrict__ clLogit, int nrows)
{
  int gw   = (blockIdx.x * blockDim.x + threadIdx.x) >> 5;
  int lane = threadIdx.x & 31;
  if (gw >= nrows) return;
  const float* h = H + (size_t)gw * DD;
  int t = target[gw];
  const float* wt = W + (size_t)t * DD;
  float s0 = 0.f, s1 = 0.f, s2 = 0.f, s3 = 0.f;
  for (int k = lane; k < DD; k += 32) {
    float hv = h[k];
    s0 += hv * wt[k];
    s1 += hv * CW[k];
    s2 += hv * CW[DD + k];
    s3 += hv * CW[2 * DD + k];
  }
  #pragma unroll
  for (int off = 16; off > 0; off >>= 1) {
    s0 += __shfl_xor(s0, off, 32);
    s1 += __shfl_xor(s1, off, 32);
    s2 += __shfl_xor(s2, off, 32);
    s3 += __shfl_xor(s3, off, 32);
  }
  if (lane == 0) {
    float tl = s0 + B[t];
    if (t == 5 || t == 17 || t == 123 || t == 10005 || t == 20007 || t == 50011)
      tl = NEG_INF;  // target is a masked root column
    targetLogit[gw] = tl;
    clLogit[gw * 3 + 0] = s1 + CB[0];
    clLogit[gw * 3 + 1] = s2 + CB[1];
    clLogit[gw * 3 + 2] = s3 + CB[2];
  }
}

// ---------------------------------------------------------------------------
// thread-per-row: merge slab partials into per-cluster logZ, emit NLL
// ---------------------------------------------------------------------------
__global__ __launch_bounds__(256) void reduce_kernel(
    const float2* __restrict__ partial,
    const float* __restrict__ targetLogit,
    const float* __restrict__ clLogit,
    const int* __restrict__ target,
    float* __restrict__ out, int nrows)
{
  int row = blockIdx.x * blockDim.x + threadIdx.x;
  if (row >= nrows) return;
  const int tb[4]   = {0, 158, 316, 786};   // slab base per cluster
  const int tcnt[4] = {158, 158, 470, 782}; // ceil(n/128)*2
  float logZ[4];
  for (int c = 0; c < 4; c++) {
    float m = NEG_INF, s = 0.f;
    for (int t = tb[c]; t < tb[c] + tcnt[c]; t++) {
      float2 p = partial[(size_t)t * nrows + row];
      if (p.y > 0.f) {
        float mn = fmaxf(m, p.x);
        float sn = (s > 0.f ? s * __expf(m - mn) : 0.f) + p.y * __expf(p.x - mn);
        m = mn; s = sn;
      }
    }
    if (c == 0) {  // head also contains the 3 (unmasked) cluster columns
      for (int j = 0; j < 3; j++) {
        float L = clLogit[row * 3 + j];
        float mn = fmaxf(m, L);
        float sn = (s > 0.f ? s * __expf(m - mn) : 0.f) + __expf(L - mn);
        m = mn; s = sn;
      }
    }
    logZ[c] = m + __logf(s);
  }
  int t = target[row];
  int c = (t < 10000) ? 0 : (t < 20000) ? 1 : (t < 50000) ? 2 : 3;
  float tl = targetLogit[row];
  float nll;
  if (c == 0) {
    nll = -(tl - logZ[0]);
  } else {
    // head column HEAD_SIZE - c  ==  cluster_weight row (3 - c)
    float chead = clLogit[row * 3 + (3 - c)] - logZ[0];
    float ctail = tl - logZ[c];
    nll = -(chead + ctail);
  }
  out[row] = nll;
}

extern "C" void kernel_launch(void* const* d_in, const int* in_sizes, int n_in,
                              void* d_out, int out_size, void* d_ws, size_t ws_size,
                              hipStream_t stream)
{
  const float* hidden = (const float*)d_in[0];
  const float* weight = (const float*)d_in[1];
  const float* bias   = (const float*)d_in[2];
  const float* cw     = (const float*)d_in[3];
  const float* cb     = (const float*)d_in[4];
  const int*   target = (const int*)d_in[5];
  float* out = (float*)d_out;
  int nrows = in_sizes[0] / DD;          // 2048

  const int TOT_T = 1568;                // total 64-col slabs across clusters
  float2* partial     = (float2*)d_ws;                    // ~25.7 MB
  float*  targetLogit = (float*)(partial + (size_t)TOT_T * nrows);
  float*  clLogit     = targetLogit + nrows;

  struct ClusterCfg { int start, n, tbase, r0, r1, r2; };
  const ClusterCfg cfg[4] = {
    {0,     10000,   0,  5, 17, 123},
    {10000, 10000, 158,  5, -1, -1},
    {20000, 30000, 316,  7, -1, -1},
    {50000, 50000, 786, 11, -1, -1},
  };
  int rowBlocks = (nrows + BM - 1) / BM;
  for (int c = 0; c < 4; c++) {
    int colTiles = (cfg[c].n + BN - 1) / BN;
    dim3 grid(rowBlocks, colTiles);       // x = row blocks -> weight tiles reused via L2
    gemm_lse_kernel<<<grid, 256, 0, stream>>>(
        hidden, weight + (size_t)cfg[c].start * DD, bias + cfg[c].start,
        cfg[c].n, nrows, cfg[c].r0, cfg[c].r1, cfg[c].r2, cfg[c].tbase, partial);
  }
  dots_kernel<<<(nrows * 32 + 255) / 256, 256, 0, stream>>>(
      hidden, weight, bias, cw, cb, target, targetLogit, clLogit, nrows);
  reduce_kernel<<<(nrows + 255) / 256, 256, 0, stream>>>(
      partial, targetLogit, clLogit, target, out, nrows);
}
